// TCAttention_74921409511526
// MI455X (gfx1250) — matrix operand order repaired and checked by reading, not compiled, not run on hardware
//
#include <hip/hip_runtime.h>
#include <hip/hip_bf16.h>

// ---------------------------------------------------------------------------
// Problem constants (match reference): B=2, T=2048, C=2048, H=32, dh=64
// ---------------------------------------------------------------------------
#define BB 2
#define TT 2048
#define CC 2048
#define HH 32
#define DH 64
#define FRAME_SIZE 576
#define MTOT (BB * TT)          // 4096 rows in all GEMMs

typedef __attribute__((ext_vector_type(16))) _Float16 v16h;
typedef __attribute__((ext_vector_type(8)))  _Float16 v8h;
typedef __attribute__((ext_vector_type(8)))  float    v8f;
typedef __attribute__((ext_vector_type(4)))  unsigned int u32x4;
typedef __attribute__((ext_vector_type(8)))  int          i32x8;
typedef __attribute__((ext_vector_type(4)))  int          i32x4;

// ---------------------------------------------------------------------------
// WMMA wrapper: D(f32 16x16) = A(f16 16x32) * B(f16 32x16) + C
// ---------------------------------------------------------------------------
__device__ __forceinline__ v8f wmma_f16(v16h a, v16h b, v8f c) {
    return __builtin_amdgcn_wmma_f32_16x16x32_f16(
        /*neg_a=*/false, a, /*neg_b=*/false, b,
        /*c_mod=*/(short)0, c, /*reuse_a=*/false, /*reuse_b=*/false);
}

// A fragment (16x32, f16), contiguous k order used by this code (B matches):
//   lane = {m = lane&15, h = lane>>4}
//   elems 0..15 : K = k0 + 16h .. +15
__device__ __forceinline__ v16h load_a_frag(const _Float16* base, int ld, int lane) {
    int m = lane & 15;
    int h = lane >> 4;
    const _Float16* p = base + (size_t)m * ld + 16 * h;
    union { v16h v; v8h half2[2]; } u;
    u.half2[0] = *(const v8h*)(p);
    u.half2[1] = *(const v8h*)(p + 8);
    return u.v;
}

// B fragment (32x16, f16), contiguous k order used by this code (A matches):
//   lane = {n = lane&15, h = lane>>4}; elems 0..15 : K = k0 + 16h .. +15
__device__ __forceinline__ v16h load_b_frag(const _Float16* base, int ld, int lane) {
    int n = lane & 15;
    int h = lane >> 4;
    return *(const v16h*)(base + (size_t)n * ld + 16 * h);
}

// ---------------------------------------------------------------------------
// Tensor Data Mover: 2-D tile (f16) global -> LDS.
// D# built per CDNA5 ISA (groups 2/3 zero => 2-D tile).
// Group0: count=1 | lds_addr | global_addr | type=2. Group1: data_size=2B,
// tensor dims (OOB), tile dims, row stride. workgroup_mask=0 (no cluster).
// This toolchain exposes the 6-arg builtin (extra int32x8 group, then cpol).
// ---------------------------------------------------------------------------
__device__ __forceinline__ void tdm_load_tile_2d(void* lds_dst, const void* gsrc,
                                                 unsigned tile_k, unsigned tile_rows,
                                                 unsigned row_stride_elems) {
    unsigned long long ga = (unsigned long long)(uintptr_t)gsrc;
    unsigned lds = (unsigned)(uintptr_t)lds_dst;   // low 32 bits = LDS byte offset
    u32x4 g0;
    g0[0] = 1u;                                    // count=1 (valid), no gather
    g0[1] = lds;                                   // lds_addr
    g0[2] = (unsigned)ga;                          // global_addr[31:0]
    g0[3] = (unsigned)(ga >> 32) | 0x80000000u;    // global_addr[56:32] | type=2
    unsigned td0 = row_stride_elems;               // tensor_dim0 (in-bounds)
    unsigned td1 = 1u << 20;                       // tensor_dim1 (in-bounds)
    i32x8 g1;
    g1[0] = (int)(1u << 16);                       // data_size = 2 bytes
    g1[1] = (int)((td0 & 0xFFFFu) << 16);          // tensor_dim0[15:0]
    g1[2] = (int)((td0 >> 16) | ((td1 & 0xFFFFu) << 16));
    g1[3] = (int)((td1 >> 16) | (tile_k << 16));   // tile_dim0 = tile_k
    g1[4] = (int)tile_rows;                        // tile_dim1
    g1[5] = (int)row_stride_elems;                 // tensor_dim0_stride[31:0]
    g1[6] = 0;
    g1[7] = 0;
    i32x4 z4 = {0, 0, 0, 0};
    i32x8 z8 = {0, 0, 0, 0, 0, 0, 0, 0};
    __builtin_amdgcn_tensor_load_to_lds(g0, g1, z4, z4, z8, 0);
}

// ---------------------------------------------------------------------------
// Kernel 0: fp32 -> f16 conversion (X and weights)
// ---------------------------------------------------------------------------
__global__ void __launch_bounds__(256) cvt_f32_to_f16(const float* __restrict__ in,
                                                      _Float16* __restrict__ out, int n) {
    for (int i = blockIdx.x * blockDim.x + threadIdx.x; i < n; i += gridDim.x * blockDim.x)
        out[i] = (_Float16)in[i];
}

// ---------------------------------------------------------------------------
// Kernel 1: projection GEMM  Y[m,n] = sum_k Xh[m,k]*Wh[n,k] + b[n] with
// optional fused dual-RoPE. Block = 128 threads (4 waves); wave tile =
// 32(M) x 64(N); the block's shared 64x32 W tile is TDM-prefetched into
// double-buffered LDS one K-step ahead.  grid = (C/64, M/128)
// ---------------------------------------------------------------------------
__global__ void __launch_bounds__(128) gemm_qkv_rope(
    const _Float16* __restrict__ Xh, const _Float16* __restrict__ Wh,
    const float* __restrict__ bias, _Float16* __restrict__ Out, int doRope) {

    __shared__ __align__(32) _Float16 Wbuf[2][64 * 32];

    const int lane = threadIdx.x & 31;
    const int wv   = threadIdx.x >> 5;
    const int m0   = (blockIdx.y * 4 + wv) * 32;
    const int n0   = blockIdx.x * 64;          // head-aligned (n0 = h*64)

    v8f acc[2][4] = {};

    const _Float16* wtile = Wh + (size_t)n0 * CC;
    if (wv == 0) tdm_load_tile_2d(&Wbuf[0][0], wtile, 32, 64, CC);

    const _Float16* arow = Xh + (size_t)m0 * CC;
    for (int k0 = 0; k0 < CC; k0 += 32) {
        const int cur = (k0 >> 5) & 1;
        if (wv == 0) __builtin_amdgcn_s_wait_tensorcnt((short)0);
        __syncthreads();                                    // buf[cur] ready
        if (wv == 0 && (k0 + 32) < CC)                      // prefetch next tile
            tdm_load_tile_2d(&Wbuf[cur ^ 1][0], wtile + k0 + 32, 32, 64, CC);

        __builtin_prefetch(arow + k0 + 512, 0, 1);          // global_prefetch_b8
        v16h a0 = load_a_frag(arow + k0, CC, lane);
        v16h a1 = load_a_frag(arow + (size_t)16 * CC + k0, CC, lane);
        #pragma unroll
        for (int jn = 0; jn < 4; ++jn) {
            v16h b = load_b_frag(&Wbuf[cur][16 * jn * 32], 32, lane);
            acc[0][jn] = wmma_f16(a0, b, acc[0][jn]);
            acc[1][jn] = wmma_f16(a1, b, acc[1][jn]);
        }
        __syncthreads();                                    // done reading buf[cur]
    }

    // Epilogue: C/D layout -> row = base + e + 8*(lane>=16), col = n0+16jn+(lane&15)
    const int nIdx = lane & 15;
    const int mAdd = (lane >> 4) * 8;

    if (doRope) {
        #pragma unroll
        for (int rg = 0; rg < 2; ++rg)
            #pragma unroll
            for (int jn = 0; jn < 2; ++jn) {
                int d = 16 * jn + nIdx;                     // 0..31 == theta index
                float theta = __powf(10000.0f, -2.0f * (float)d / (float)DH);
                float b1 = bias[n0 + 16 * jn + nIdx];
                float b2 = bias[n0 + 16 * jn + 32 + nIdx];
                #pragma unroll
                for (int e = 0; e < 8; ++e) {
                    int m = m0 + 16 * rg + e + mAdd;
                    int t = m & (TT - 1);                   // m % T
                    int tmp = (t / FRAME_SIZE) >> 1;        // floor(frame * 0.5)
                    float ang = (float)(t + tmp) * theta;   // fused dual-RoPE angle
                    float sn, cs;
                    __sincosf(ang, &sn, &cs);
                    float x1 = acc[rg][jn][e] + b1;
                    float x2 = acc[rg][jn + 2][e] + b2;
                    acc[rg][jn][e]     = x1 * cs - x2 * sn;
                    acc[rg][jn + 2][e] = x2 * cs + x1 * sn;
                }
            }
    } else {
        #pragma unroll
        for (int rg = 0; rg < 2; ++rg)
            #pragma unroll
            for (int jn = 0; jn < 4; ++jn) {
                float bn = bias[n0 + 16 * jn + nIdx];
                #pragma unroll
                for (int e = 0; e < 8; ++e) acc[rg][jn][e] += bn;
            }
    }

    #pragma unroll
    for (int rg = 0; rg < 2; ++rg)
        #pragma unroll
        for (int jn = 0; jn < 4; ++jn)
            #pragma unroll
            for (int e = 0; e < 8; ++e)
                Out[(size_t)(m0 + 16 * rg + e + mAdd) * CC + n0 + 16 * jn + nIdx] =
                    (_Float16)acc[rg][jn][e];
}

// ---------------------------------------------------------------------------
// Kernel 2: causal flash attention, one block per (b, h, 64-query tile).
// Block = 128 threads (4 waves); wave owns 16 query rows, full dh=64.
// K tile is TDM-loaded into LDS, V tile is thread-transposed in parallel.
// ---------------------------------------------------------------------------
#define VTS 80   // Vt row stride in halves (32B aligned, bank-spread)

__global__ void __launch_bounds__(128) flash_attn(
    const _Float16* __restrict__ Qh, const _Float16* __restrict__ Kh,
    const _Float16* __restrict__ Vh, _Float16* __restrict__ Oh) {

    __shared__ __align__(32) _Float16 Ks[64 * 64];     // [key][d]  row-major (TDM)
    __shared__ __align__(32) _Float16 Vt[64 * VTS];    // [d][key]  transposed
    __shared__ __align__(32) float    Sb[64 * 64];     // scores fp32
    __shared__ __align__(32) _Float16 Pb[64 * 64];     // probs f16
    __shared__ float alphaB[64];
    __shared__ float linvB[64];

    const int lane = threadIdx.x & 31;
    const int wv   = threadIdx.x >> 5;
    const int q0   = blockIdx.x * 64;
    const int bh   = blockIdx.y;
    const int b    = bh >> 5;
    const int h    = bh & (HH - 1);
    const float scale = 0.125f;                        // 1/sqrt(64)

    // Preload this wave's Q fragments (two K-steps over dh=64)
    const _Float16* qbase = Qh + ((size_t)(b * TT + q0 + 16 * wv)) * CC + h * DH;
    v16h aq0 = load_a_frag(qbase + 0,  CC, lane);
    v16h aq1 = load_a_frag(qbase + 32, CC, lane);

    v8f o[4] = {};
    float mrow = -1e30f, lrow = 0.0f;                  // lanes 0..15 only

    const int nIdx = lane & 15;
    const int mAdd = (lane >> 4) * 8;

    for (int j = 0; j <= (int)blockIdx.x; ++j) {
        const size_t tile0 = ((size_t)(b * TT + j * 64)) * CC + h * DH;

        // ---- K tile via Tensor Data Mover, V tile transposed by threads ----
        if (wv == 0) tdm_load_tile_2d(&Ks[0], Kh + tile0, 64, 64, CC);
        #pragma unroll
        for (int it = 0; it < 4; ++it) {
            int idx  = threadIdx.x + 128 * it;          // 512 chunks of 8 halves
            int row  = idx >> 3;
            int col8 = (idx & 7) * 8;
            v8h vvv = *(const v8h*)(Vh + tile0 + (size_t)row * CC + col8);
            #pragma unroll
            for (int e = 0; e < 8; ++e) Vt[(col8 + e) * VTS + row] = vvv[e];
        }
        if (wv == 0) __builtin_amdgcn_s_wait_tensorcnt((short)0);
        __syncthreads();

        // ---- S = Q K^T (8 WMMAs per wave) ----
        #pragma unroll
        for (int jn = 0; jn < 4; ++jn) {
            v8f s = v8f{};
            s = wmma_f16(aq0, load_b_frag(Ks + 16 * jn * 64 + 0,  64, lane), s);
            s = wmma_f16(aq1, load_b_frag(Ks + 16 * jn * 64 + 32, 64, lane), s);
            #pragma unroll
            for (int e = 0; e < 8; ++e)
                Sb[(16 * wv + e + mAdd) * 64 + 16 * jn + nIdx] = s[e] * scale;
        }
        __syncthreads();

        // ---- online softmax: one row per lane (lanes 0..15 of each wave) ----
        if (lane < 16) {
            int row  = 16 * wv + lane;
            int kmax = (j == (int)blockIdx.x) ? (row + 1) : 64;   // causal bound
            float mnew = mrow;
            for (int k = 0; k < kmax; ++k) mnew = fmaxf(mnew, Sb[row * 64 + k]);
            float alpha = __expf(mrow - mnew);
            float ssum = 0.0f;
            for (int k = 0; k < 64; ++k) {
                float p = (k < kmax) ? __expf(Sb[row * 64 + k] - mnew) : 0.0f;
                Pb[row * 64 + k] = (_Float16)p;
                ssum += p;
            }
            lrow = lrow * alpha + ssum;
            mrow = mnew;
            alphaB[row] = alpha;
        }
        __syncthreads();

        // ---- rescale O, then O += P * V (8 WMMAs per wave) ----
        #pragma unroll
        for (int jd = 0; jd < 4; ++jd)
            #pragma unroll
            for (int e = 0; e < 8; ++e) o[jd][e] *= alphaB[16 * wv + e + mAdd];

        #pragma unroll
        for (int ks = 0; ks < 2; ++ks) {
            v16h ap = load_a_frag(Pb + 16 * wv * 64 + 32 * ks, 64, lane);
            #pragma unroll
            for (int jd = 0; jd < 4; ++jd) {
                v16h bv = load_b_frag(Vt + 16 * jd * VTS + 32 * ks, VTS, lane);
                o[jd] = wmma_f16(ap, bv, o[jd]);
            }
        }
        __syncthreads();   // protect LDS tiles before next iteration
    }

    // ---- normalize and store O as f16 in [B,T,H,dh] (== [B,T,C]) ----
    if (lane < 16) linvB[16 * wv + lane] = 1.0f / lrow;
    __syncthreads();

    #pragma unroll
    for (int jd = 0; jd < 4; ++jd)
        #pragma unroll
        for (int e = 0; e < 8; ++e) {
            int row = 16 * wv + e + mAdd;
            float v = o[jd][e] * linvB[row];
            Oh[((size_t)(b * TT + q0 + row)) * CC + h * DH + 16 * jd + nIdx] = (_Float16)v;
        }
}

// ---------------------------------------------------------------------------
// Kernel 3: output projection out[m,n] = sum_k Oh[m,k]*Wo[n,k] + bo[n] (fp32),
// same TDM double-buffered mainloop as kernel 1.
// ---------------------------------------------------------------------------
__global__ void __launch_bounds__(128) gemm_out(
    const _Float16* __restrict__ Ah, const _Float16* __restrict__ Wh,
    const float* __restrict__ bias, float* __restrict__ Out) {

    __shared__ __align__(32) _Float16 Wbuf[2][64 * 32];

    const int lane = threadIdx.x & 31;
    const int wv   = threadIdx.x >> 5;
    const int m0   = (blockIdx.y * 4 + wv) * 32;
    const int n0   = blockIdx.x * 64;

    v8f acc[2][4] = {};

    const _Float16* wtile = Wh + (size_t)n0 * CC;
    if (wv == 0) tdm_load_tile_2d(&Wbuf[0][0], wtile, 32, 64, CC);

    const _Float16* arow = Ah + (size_t)m0 * CC;
    for (int k0 = 0; k0 < CC; k0 += 32) {
        const int cur = (k0 >> 5) & 1;
        if (wv == 0) __builtin_amdgcn_s_wait_tensorcnt((short)0);
        __syncthreads();
        if (wv == 0 && (k0 + 32) < CC)
            tdm_load_tile_2d(&Wbuf[cur ^ 1][0], wtile + k0 + 32, 32, 64, CC);

        __builtin_prefetch(arow + k0 + 512, 0, 1);
        v16h a0 = load_a_frag(arow + k0, CC, lane);
        v16h a1 = load_a_frag(arow + (size_t)16 * CC + k0, CC, lane);
        #pragma unroll
        for (int jn = 0; jn < 4; ++jn) {
            v16h b = load_b_frag(&Wbuf[cur][16 * jn * 32], 32, lane);
            acc[0][jn] = wmma_f16(a0, b, acc[0][jn]);
            acc[1][jn] = wmma_f16(a1, b, acc[1][jn]);
        }
        __syncthreads();
    }

    const int nIdx = lane & 15;
    const int mAdd = (lane >> 4) * 8;
    #pragma unroll
    for (int rg = 0; rg < 2; ++rg)
        #pragma unroll
        for (int jn = 0; jn < 4; ++jn) {
            float bn = bias[n0 + 16 * jn + nIdx];
            #pragma unroll
            for (int e = 0; e < 8; ++e)
                Out[(size_t)(m0 + 16 * rg + e + mAdd) * CC + n0 + 16 * jn + nIdx] =
                    acc[rg][jn][e] + bn;
        }
}

// ---------------------------------------------------------------------------
// Host-side orchestration
// ---------------------------------------------------------------------------
extern "C" void kernel_launch(void* const* d_in, const int* in_sizes, int n_in,
                              void* d_out, int out_size, void* d_ws, size_t ws_size,
                              hipStream_t stream) {
    const float* X  = (const float*)d_in[0];
    const float* Wq = (const float*)d_in[1];
    const float* bq = (const float*)d_in[2];
    const float* Wk = (const float*)d_in[3];
    const float* bk = (const float*)d_in[4];
    const float* Wv = (const float*)d_in[5];
    const float* bv = (const float*)d_in[6];
    const float* Wo = (const float*)d_in[7];
    const float* bo = (const float*)d_in[8];
    float* out = (float*)d_out;

    const size_t NX = (size_t)MTOT * CC;   // 8,388,608
    const size_t NW = (size_t)CC * CC;     // 4,194,304

    _Float16* Xh  = (_Float16*)d_ws;
    _Float16* Wqh = Xh  + NX;
    _Float16* Wkh = Wqh + NW;
    _Float16* Wvh = Wkh + NW;
    _Float16* Woh = Wvh + NW;
    _Float16* Qh  = Woh + NW;
    _Float16* Kh  = Qh  + NX;
    _Float16* Vh  = Kh  + NX;
    _Float16* Oh  = Vh  + NX;

    // 0) convert X and weights to f16
    cvt_f32_to_f16<<<1024, 256, 0, stream>>>(X,  Xh,  (int)NX);
    cvt_f32_to_f16<<<1024, 256, 0, stream>>>(Wq, Wqh, (int)NW);
    cvt_f32_to_f16<<<1024, 256, 0, stream>>>(Wk, Wkh, (int)NW);
    cvt_f32_to_f16<<<1024, 256, 0, stream>>>(Wv, Wvh, (int)NW);
    cvt_f32_to_f16<<<1024, 256, 0, stream>>>(Wo, Woh, (int)NW);

    // 1) QKV projections (+ fused dual-RoPE on Q, K)
    dim3 gGemm(CC / 64, MTOT / 128);
    gemm_qkv_rope<<<gGemm, 128, 0, stream>>>(Xh, Wqh, bq, Qh, 1);
    gemm_qkv_rope<<<gGemm, 128, 0, stream>>>(Xh, Wkh, bk, Kh, 1);
    gemm_qkv_rope<<<gGemm, 128, 0, stream>>>(Xh, Wvh, bv, Vh, 0);

    // 2) causal flash attention
    dim3 gAttn(TT / 64, BB * HH);
    flash_attn<<<gAttn, 128, 0, stream>>>(Qh, Kh, Vh, Oh);

    // 3) output projection -> fp32
    gemm_out<<<gGemm, 128, 0, stream>>>(Oh, Woh, bo, out);
}
